// GCNSkip_996432413504
// MI455X (gfx1250) — matrix operand order repaired and checked by reading, not compiled, not run on hardware
//
#include <hip/hip_runtime.h>
#include <hip/hip_bf16.h>
#include <math.h>

#define HID      128
#define N_GRAPHS 128
#define N_CLASSES 10

typedef __attribute__((ext_vector_type(16))) __bf16 v16bf;
typedef __attribute__((ext_vector_type(8)))  __bf16 v8bf;
typedef __attribute__((ext_vector_type(4)))  __bf16 v4bf;
typedef __attribute__((ext_vector_type(8)))  float  v8f;

// ---------------------------------------------------------------------------
// Weight prep: fp32 [128,128] row-major -> bf16 WMMA B-fragment layout.
// Fragment (kb, nt): lane's 16 values stored contiguously (32 bytes):
//   frag[j] = B[(kb*32 + 16*half + j)*128 + nt*16 + lc]
// ---------------------------------------------------------------------------
__global__ __launch_bounds__(256) void gcn_prep_w(const float* __restrict__ B,
                                                  __bf16* __restrict__ out)
{
  int t    = blockIdx.x * blockDim.x + threadIdx.x;   // 0..1023
  int lane = t & 31;
  int nt   = (t >> 5) & 7;
  int kb   = t >> 8;
  int half = lane >> 4, lc = lane & 15;
  __bf16* dst = out + (((size_t)(kb * 8 + nt) * 32 + lane) * 16);
  const float* src = B + (size_t)(kb * 32 + 16 * half) * HID + nt * 16 + lc;
#pragma unroll
  for (int j = 0; j < 16; ++j) dst[j] = (__bf16)src[(size_t)j * HID];
}

// fp32 row-major -> bf16 row-major (4 elems/thread)
__global__ __launch_bounds__(256) void gcn_f32_to_bf16(
    const float* __restrict__ in, __bf16* __restrict__ out, int n4)
{
  int i = blockIdx.x * blockDim.x + threadIdx.x;
  if (i >= n4) return;
  float4 f = ((const float4*)in)[i];
  v4bf o; o[0]=(__bf16)f.x; o[1]=(__bf16)f.y; o[2]=(__bf16)f.z; o[3]=(__bf16)f.w;
  ((v4bf*)out)[i] = o;
}

// ---------------------------------------------------------------------------
// WMMA GEMM: C[nrows,128] = opt_relu( A[nrows,128] @ B[128,128] + add + bias )
// A bf16 row-major; Bp prepped bf16 fragments. 8 waves/block cover 32 rows x
// 128 cols; each wave computes TWO 16x16 tiles sharing one A fragment:
// per K-step = 2 A-loads + 4 B-loads (all b128) + 2 WMMA. Compile-time
// epilogue flags keep the tail branch-free.
// ---------------------------------------------------------------------------
template <bool HAS_ADD, bool HAS_BIAS, bool RELU, bool OUT_F32, bool OUT_BF16>
__global__ __launch_bounds__(256) void gcn_gemm128(
    const __bf16* __restrict__ A, const __bf16* __restrict__ Bp,
    const float* __restrict__ bias, const float* __restrict__ add,
    float* __restrict__ Cf, __bf16* __restrict__ Cb, int nrows)
{
  const int wave = threadIdx.x >> 5;
  const int lane = threadIdx.x & 31;
  const int half = lane >> 4;
  const int lc   = lane & 15;
  const int m0   = blockIdx.x * 32 + (wave >> 2) * 16;  // row tile
  const int nt0  = (wave & 3) * 2;                      // first of 2 col tiles
  const int n0   = nt0 * 16;
  if (m0 >= nrows) return;   // wave-uniform tail exit (EXEC stays all-ones)

  const __bf16* Arow = A + (size_t)(m0 + lc) * HID;
  v8f acc0 = {}, acc1 = {};

#pragma unroll
  for (int kb = 0; kb < 4; ++kb) {
    const int k0 = kb * 32;
    // A frag: lanes 0-15 K={k0..+15}; lanes 16-31 K={k0+16..+31}
    v8bf a0 = *(const v8bf*)(Arow + k0 + half * 16);
    v8bf a1 = *(const v8bf*)(Arow + k0 + half * 16 + 8);
    v16bf a;
#pragma unroll
    for (int j = 0; j < 8; ++j) { a[j] = a0[j]; a[8 + j] = a1[j]; }
    // Two B frags: contiguous 32B per lane, adjacent tiles 512 elems apart
    const __bf16* bp = Bp + (((size_t)(kb * 8 + nt0) * 32 + lane) * 16);
    v8bf b00 = *(const v8bf*)bp;
    v8bf b01 = *(const v8bf*)(bp + 8);
    v8bf b10 = *(const v8bf*)(bp + 512);
    v8bf b11 = *(const v8bf*)(bp + 520);
    v16bf b0, b1;
#pragma unroll
    for (int j = 0; j < 8; ++j) {
      b0[j] = b00[j]; b0[8 + j] = b01[j];
      b1[j] = b10[j]; b1[8 + j] = b11[j];
    }
    acc0 = __builtin_amdgcn_wmma_f32_16x16x32_bf16(
        false, a, false, b0, (short)0, acc0, false, false);
    acc1 = __builtin_amdgcn_wmma_f32_16x16x32_bf16(
        false, a, false, b1, (short)0, acc1, false, false);
  }

  // C/D layout: row = m0 + v + 8*half, col = n0 + lc (tile0) / +16 (tile1)
#pragma unroll
  for (int v = 0; v < 8; ++v) {
    const int row = m0 + v + 8 * half;
    const size_t base = (size_t)row * HID + n0 + lc;
    float v0 = acc0[v], v1 = acc1[v];
    if (HAS_ADD)  { v0 += add[base]; v1 += add[base + 16]; }
    if (HAS_BIAS) { v0 += bias[n0 + lc]; v1 += bias[n0 + lc + 16]; }
    if (RELU)     { v0 = fmaxf(v0, 0.0f); v1 = fmaxf(v1, 0.0f); }
    if (OUT_F32)  { Cf[base] = v0; Cf[base + 16] = v1; }
    if (OUT_BF16) { Cb[base] = (__bf16)v0; Cb[base + 16] = (__bf16)v1; }
  }
}

// ---------------------------------------------------------------------------
// GCN normalization helpers
// ---------------------------------------------------------------------------
__global__ void gcn_degree(const int* __restrict__ dst, float* __restrict__ deg,
                           int nEdges)
{
  int e = blockIdx.x * blockDim.x + threadIdx.x;
  if (e < nEdges) unsafeAtomicAdd(&deg[dst[e]], 1.0f);
}

__global__ void gcn_dinv(float* __restrict__ deg, int n)
{
  int i = blockIdx.x * blockDim.x + threadIdx.x;
  if (i < n) {
    float d = deg[i];
    deg[i] = (d > 0.0f) ? rsqrtf(fmaxf(d, 1.0f)) : 0.0f;
  }
}

__global__ void gcn_norm(const int* __restrict__ src, const int* __restrict__ dst,
                         const float* __restrict__ dinv, float* __restrict__ norm,
                         int nEdges)
{
  int e = blockIdx.x * blockDim.x + threadIdx.x;
  if (e < nEdges) norm[e] = dinv[src[e]] * dinv[dst[e]];
}

// ---------------------------------------------------------------------------
// Edge scatter: agg[dst] += norm[e] * hw_bf16[src]; 32 threads/edge, 4 bf16
// (8B) per thread. f32 atomics resolve in the L2-resident agg buffer.
// ---------------------------------------------------------------------------
__global__ __launch_bounds__(256) void gcn_scatter(
    const __bf16* __restrict__ hw, const int* __restrict__ src,
    const int* __restrict__ dst, const float* __restrict__ norm,
    float* __restrict__ agg, int nEdges)
{
  long long tid = (long long)blockIdx.x * blockDim.x + threadIdx.x;
  int e = (int)(tid >> 5);
  if (e >= nEdges) return;
  int q = ((int)tid & 31) * 4;
  float nv = norm[e];
  int s = src[e], d = dst[e];
  v4bf m = *(const v4bf*)(hw + (size_t)s * HID + q);
  float* ap = agg + (size_t)d * HID + q;
  unsafeAtomicAdd(ap + 0, (float)m[0] * nv);
  unsafeAtomicAdd(ap + 1, (float)m[1] * nv);
  unsafeAtomicAdd(ap + 2, (float)m[2] * nv);
  unsafeAtomicAdd(ap + 3, (float)m[3] * nv);
}

// ---------------------------------------------------------------------------
// Global mean pool (reads fp32 `last`)
// ---------------------------------------------------------------------------
__global__ __launch_bounds__(256) void gcn_pool(
    const float* __restrict__ h, const int* __restrict__ batch,
    float* __restrict__ pooled, float* __restrict__ cnt, int nNodes)
{
  long long tid = (long long)blockIdx.x * blockDim.x + threadIdx.x;
  int n = (int)(tid >> 5);
  if (n >= nNodes) return;
  int q = ((int)tid & 31) * 4;
  int g = batch[n];
  float4 m = *(const float4*)(h + (size_t)n * HID + q);
  float* pp = pooled + (size_t)g * HID + q;
  unsafeAtomicAdd(pp + 0, m.x);
  unsafeAtomicAdd(pp + 1, m.y);
  unsafeAtomicAdd(pp + 2, m.z);
  unsafeAtomicAdd(pp + 3, m.w);
  if (q == 0) unsafeAtomicAdd(&cnt[g], 1.0f);
}

__global__ void gcn_pool_div(float* __restrict__ pooled,
                             const float* __restrict__ cnt)
{
  int i = blockIdx.x * blockDim.x + threadIdx.x;
  if (i < N_GRAPHS * HID) pooled[i] /= fmaxf(cnt[i >> 7], 1.0f);
}

// ---------------------------------------------------------------------------
// Head: logits = g @ lin2_w + lin2_b, then log_softmax. One wave per graph.
// ---------------------------------------------------------------------------
__global__ __launch_bounds__(32) void gcn_head(
    const float* __restrict__ g, const float* __restrict__ w2,
    const float* __restrict__ b2, float* __restrict__ out)
{
  int graph = blockIdx.x;
  int j = threadIdx.x;
  __shared__ float logits[N_CLASSES];
  __shared__ float red[2];
  if (j < N_CLASSES) {
    float s = b2[j];
    const float* gr = g + (size_t)graph * HID;
#pragma unroll 8
    for (int k = 0; k < HID; ++k) s += gr[k] * w2[k * N_CLASSES + j];
    logits[j] = s;
  }
  __syncthreads();
  if (j == 0) {
    float m = logits[0];
    for (int k = 1; k < N_CLASSES; ++k) m = fmaxf(m, logits[k]);
    float lse = 0.0f;
    for (int k = 0; k < N_CLASSES; ++k) lse += __expf(logits[k] - m);
    red[0] = m; red[1] = __logf(lse);
  }
  __syncthreads();
  if (j < N_CLASSES) out[(size_t)graph * N_CLASSES + j] = logits[j] - red[0] - red[1];
}

// ---------------------------------------------------------------------------
extern "C" void kernel_launch(void* const* d_in, const int* in_sizes, int n_in,
                              void* d_out, int out_size, void* d_ws, size_t ws_size,
                              hipStream_t stream)
{
  const float* x      = (const float*)d_in[0];
  const int*   ei     = (const int*)d_in[1];
  const int*   batch  = (const int*)d_in[2];
  const float* W1     = (const float*)d_in[3];
  const float* R1     = (const float*)d_in[4];
  const float* b1     = (const float*)d_in[5];
  const float* Wc     = (const float*)d_in[6];
  const float* Rc     = (const float*)d_in[7];
  const float* bc     = (const float*)d_in[8];
  const float* lin1_w = (const float*)d_in[9];
  const float* lin1_b = (const float*)d_in[10];
  const float* lin2_w = (const float*)d_in[11];
  const float* lin2_b = (const float*)d_in[12];

  const int N = in_sizes[0] / HID;       // 50000
  const int E = in_sizes[1] / 2;         // 600000
  const int* src = ei;
  const int* dst = ei + E;

  float* out  = (float*)d_out;
  float* last = out + N_GRAPHS * N_CLASSES;     // [N,128] fp32

  // ---- workspace carve-up: f32 regions first, then bf16 (~70 MB) ----
  const size_t NH = (size_t)N * HID;            // 6.4M elems
  float*  AGG     = (float*)d_ws;               // [N,128] f32
  float*  norm    = AGG + NH;                   // [E]
  float*  dinv    = norm + E;                   // [N]
  float*  pooled  = dinv + N;                   // [128,128]
  float*  cnt     = pooled + N_GRAPHS * HID;    // [128]
  float*  gbuf    = cnt + N_GRAPHS;             // [128,128]
  __bf16* xbf     = (__bf16*)(gbuf + N_GRAPHS * HID);
  __bf16* B0      = xbf + NH;                   // bf16 ping
  __bf16* B1      = B0 + NH;                    // bf16 pong
  __bf16* pbf     = B1 + NH;                    // pooled bf16 [128,128]
  __bf16* wprep   = pbf + N_GRAPHS * HID;       // 9 x 16384 bf16 prepped weights

  const size_t WSZ = (size_t)HID * HID;         // 16384
  __bf16* pW[4] = { wprep,           wprep + 2*WSZ, wprep + 4*WSZ, wprep + 6*WSZ };
  __bf16* pR[4] = { wprep + 1*WSZ,   wprep + 3*WSZ, wprep + 5*WSZ, wprep + 7*WSZ };
  __bf16* pL1   = wprep + 8*WSZ;

  const int TB = 256;
  dim3 blk(TB);
  const int gemmBlocks = (N + 31) / 32;         // 1563 (32 rows per block)
  const long long scatterT = (long long)E * 32;
  const long long poolT    = (long long)N * 32;

  // ---- one-time preps: weights to frag layout, x to bf16 ----
  gcn_prep_w<<<4, blk, 0, stream>>>(W1, pW[0]);
  gcn_prep_w<<<4, blk, 0, stream>>>(R1, pR[0]);
  for (int i = 0; i < 3; ++i) {
    gcn_prep_w<<<4, blk, 0, stream>>>(Wc + (size_t)i * WSZ, pW[i + 1]);
    gcn_prep_w<<<4, blk, 0, stream>>>(Rc + (size_t)i * WSZ, pR[i + 1]);
  }
  gcn_prep_w<<<4, blk, 0, stream>>>(lin1_w, pL1);
  gcn_f32_to_bf16<<<(int)((NH / 4 + TB - 1) / TB), blk, 0, stream>>>(
      x, xbf, (int)(NH / 4));

  // ---- GCN normalization coefficients ----
  hipMemsetAsync(dinv, 0, (size_t)N * sizeof(float), stream);
  gcn_degree<<<(E + TB - 1) / TB, blk, 0, stream>>>(dst, dinv, E);
  gcn_dinv<<<(N + TB - 1) / TB, blk, 0, stream>>>(dinv, N);
  gcn_norm<<<(E + TB - 1) / TB, blk, 0, stream>>>(src, dst, dinv, norm, E);

  // ---- 4 GCNSkip layers ----
  // plan: (input_bf, hw_bf, out_bf, out_f32)
  const __bf16* lin[4] = { xbf, B1, B0, B1 };
  __bf16*       hwb[4] = { B0,  B0, B1, B0 };
  __bf16*       lob[4] = { B1,  B0, B1, nullptr };
  const float*  bl[4]  = { b1, bc, bc + HID, bc + 2 * HID };

  for (int l = 0; l < 4; ++l) {
    hipMemsetAsync(AGG, 0, NH * sizeof(float), stream);
    // hw = in @ W  (bf16 out, no epilogue)
    gcn_gemm128<false, false, false, false, true>
        <<<gemmBlocks, blk, 0, stream>>>(
            lin[l], pW[l], nullptr, nullptr, nullptr, hwb[l], N);
    // agg[dst] += norm * hw[src]
    gcn_scatter<<<(int)((scatterT + TB - 1) / TB), blk, 0, stream>>>(
        hwb[l], src, dst, norm, AGG, E);
    // out = relu(agg + in @ R + b)
    if (l < 3) {
      gcn_gemm128<true, true, true, false, true>
          <<<gemmBlocks, blk, 0, stream>>>(
              lin[l], pR[l], bl[l], AGG, nullptr, lob[l], N);
    } else {
      gcn_gemm128<true, true, true, true, false>
          <<<gemmBlocks, blk, 0, stream>>>(
              lin[l], pR[l], bl[l], AGG, last, nullptr, N);
    }
  }

  // ---- global mean pool ----
  hipMemsetAsync(pooled, 0, (size_t)(N_GRAPHS * HID + N_GRAPHS) * sizeof(float),
                 stream);
  gcn_pool<<<(int)((poolT + TB - 1) / TB), blk, 0, stream>>>(
      last, batch, pooled, cnt, N);
  gcn_pool_div<<<(N_GRAPHS * HID + TB - 1) / TB, blk, 0, stream>>>(pooled, cnt);

  // ---- MLP head ----
  gcn_f32_to_bf16<<<(N_GRAPHS * HID / 4 + TB - 1) / TB, blk, 0, stream>>>(
      pooled, pbf, N_GRAPHS * HID / 4);
  gcn_gemm128<false, true, true, true, false>
      <<<(N_GRAPHS + 31) / 32, blk, 0, stream>>>(
          pbf, pL1, lin1_b, nullptr, gbuf, nullptr, N_GRAPHS);
  gcn_head<<<N_GRAPHS, dim3(32), 0, stream>>>(gbuf, lin2_w, lin2_b, out);
}
